// MultiHeadAttention_52759378264454
// MI455X (gfx1250) — matrix operand order repaired and checked by reading, not compiled, not run on hardware
//
#include <hip/hip_runtime.h>
#include <hip/hip_bf16.h>

#define DEV __device__ __forceinline__

typedef __attribute__((ext_vector_type(16))) __bf16 bf16x16;
typedef __attribute__((ext_vector_type(8)))  __bf16 bf16x8;
typedef __attribute__((ext_vector_type(8)))  float  f32x8;

constexpr int T_ = 1024, B_ = 8, D_ = 256, H_ = 8, E_ = 512;
constexpr int HE   = H_ * E_;   // 4096
constexpr int NROW = T_ * B_;   // 8192

// ---------------- WMMA fragment helpers (layouts per CDNA5 ISA 7.12.2) ----
// A 16x32 bf16, row-major source: lane (m, half): VGPR0-3 = K half*8..+7,
// VGPR4-7 = K 16+half*8..+7  -> two 16B loads.
DEV bf16x16 load_frag_a(const __bf16* __restrict__ a, int lda, int k) {
  const int lane = threadIdx.x & 31;
  const int m = lane & 15, h = lane >> 4;
  const __bf16* p = a + (size_t)m * lda + k + h * 8;
  bf16x8 lo = *(const bf16x8*)(p);
  bf16x8 hi = *(const bf16x8*)(p + 16);
  return __builtin_shufflevector(lo, hi, 0,1,2,3,4,5,6,7,8,9,10,11,12,13,14,15);
}

// B 32x16 bf16 supplied as B^T row-major (row n contiguous over K):
// lane (n, half) holds K = half*8..+7 and 16+half*8..+7 of column n -> two 16B loads.
DEV bf16x16 load_frag_b(const __bf16* __restrict__ bt, int ldb, int k) {
  const int lane = threadIdx.x & 31;
  const int n = lane & 15, h = lane >> 4;
  const __bf16* p = bt + (size_t)n * ldb + k + h * 8;
  bf16x8 lo = *(const bf16x8*)(p);
  bf16x8 hi = *(const bf16x8*)(p + 16);
  return __builtin_shufflevector(lo, hi, 0,1,2,3,4,5,6,7,8,9,10,11,12,13,14,15);
}

DEV f32x8 wmma_bf16(bf16x16 a, bf16x16 b, f32x8 c) {
  return __builtin_amdgcn_wmma_f32_16x16x32_bf16(
      /*neg_a=*/false, a, /*neg_b=*/false, b,
      /*c_mod=*/(short)0, c, /*reuse_a=*/false, /*reuse_b=*/false);
}

struct Acc2x4 { f32x8 m[2][4]; };

// 32x64 output per wave: 2 A frags x 4 B frags -> 8 WMMA per 12 b128 loads
// (~22 FLOP/byte from cache vs 16 for a 2x2 tile).
DEV void gemm_core_2x4(const __bf16* __restrict__ A, int lda,
                       const __bf16* __restrict__ Bt, int ldb,
                       int K, Acc2x4& acc) {
  const int lane = threadIdx.x & 31;
  const int m = lane & 15;
  for (int k = 0; k < K; k += 32) {
    if (k + 32 < K) {  // gfx1250 global_prefetch_b8 of next K slice
      __builtin_prefetch(A  + (size_t)m * lda + k + 32, 0, 1);
      __builtin_prefetch(Bt + (size_t)m * ldb + k + 32, 0, 1);
    }
    bf16x16 a0 = load_frag_a(A, lda, k);
    bf16x16 a1 = load_frag_a(A + (size_t)16 * lda, lda, k);
#pragma unroll
    for (int tj = 0; tj < 4; ++tj) {
      bf16x16 b = load_frag_b(Bt + (size_t)(16 * tj) * ldb, ldb, k);
      acc.m[0][tj] = wmma_bf16(a0, b, acc.m[0][tj]);
      acc.m[1][tj] = wmma_bf16(a1, b, acc.m[1][tj]);
    }
  }
}

// Wave tile origin inside a 64x128 block tile (4 waves arranged 2x2).
DEV void wave_origin(int& M0, int& N0) {
  const int w = threadIdx.x >> 5;
  M0 = blockIdx.x * 64  + (w >> 1) * 32;
  N0 = blockIdx.y * 128 + (w & 1) * 64;
}

// ---------------- kernels --------------------------------------------------

__global__ __launch_bounds__(256) void k_cast(const float* __restrict__ src,
                                              __bf16* __restrict__ dst, int n) {
  for (int i = blockIdx.x * 256 + threadIdx.x; i < n; i += gridDim.x * 256)
    dst[i] = (__bf16)src[i];
}

// q/k/v projection: [NROW,D] x [HE,D]^T -> [NROW,HE], +bias, optional scale,
// scattered to [h][b][t][e] (vtrans=0) or [h][b][e][t] (vtrans=1).
__global__ __launch_bounds__(128) void k_qkv(const __bf16* __restrict__ X,
                                             const __bf16* __restrict__ W,
                                             const float* __restrict__ bias,
                                             __bf16* __restrict__ dst,
                                             float scale, int vtrans) {
  int M0, N0; wave_origin(M0, N0);
  Acc2x4 acc = {};
  gemm_core_2x4(X + (size_t)M0 * D_, D_, W + (size_t)N0 * D_, D_, D_, acc);

  const int lane = threadIdx.x & 31;
  const int n = lane & 15, h = lane >> 4;
#pragma unroll
  for (int ti = 0; ti < 2; ++ti)
#pragma unroll
    for (int tj = 0; tj < 4; ++tj)
#pragma unroll
      for (int r = 0; r < 8; ++r) {
        const int row = M0 + ti * 16 + h * 8 + r;   // = t*B + b
        const int col = N0 + tj * 16 + n;           // = hh*E + e
        const float v = (acc.m[ti][tj][r] + bias[col]) * scale;
        const int t = row >> 3, b = row & 7;
        const int hh = col >> 9, e = col & 511;
        const size_t slice = (size_t)(hh * B_ + b);
        if (vtrans) dst[(slice * E_ + e) * T_ + t] = (__bf16)v;   // V^T
        else        dst[(slice * T_ + t) * E_ + e] = (__bf16)v;   // Q,K
      }
}

// S = Qhb[T,E] x Khb[T,E]^T (scale pre-folded into Q), fp32 out [T,T]
__global__ __launch_bounds__(128) void k_scores(const __bf16* __restrict__ Qhb,
                                                const __bf16* __restrict__ Khb,
                                                float* __restrict__ S) {
  int M0, N0; wave_origin(M0, N0);
  Acc2x4 acc = {};
  gemm_core_2x4(Qhb + (size_t)M0 * E_, E_, Khb + (size_t)N0 * E_, E_, E_, acc);

  const int lane = threadIdx.x & 31;
  const int n = lane & 15, h = lane >> 4;
#pragma unroll
  for (int ti = 0; ti < 2; ++ti)
#pragma unroll
    for (int tj = 0; tj < 4; ++tj)
#pragma unroll
      for (int r = 0; r < 8; ++r)
        S[(size_t)(M0 + ti * 16 + h * 8 + r) * T_ + (N0 + tj * 16 + n)] =
            acc.m[ti][tj][r];
}

// Softmax over the QUERY axis t (reference softmaxes axis=2 of [H,B,Tq,Tk]):
// one block per column s, reduce over all t, emit bf16 P[t][s].
__global__ __launch_bounds__(256) void k_softmax_col(const float* __restrict__ S,
                                                     __bf16* __restrict__ P) {
  const int s = blockIdx.x;
  __shared__ float red[256];
  const int tid = threadIdx.x;

  float mx = -3.402823466e38f;
  for (int t = tid; t < T_; t += 256) mx = fmaxf(mx, S[(size_t)t * T_ + s]);
  red[tid] = mx; __syncthreads();
  for (int off = 128; off > 0; off >>= 1) {
    if (tid < off) red[tid] = fmaxf(red[tid], red[tid + off]);
    __syncthreads();
  }
  mx = red[0]; __syncthreads();

  float sum = 0.f;
  for (int t = tid; t < T_; t += 256) sum += __expf(S[(size_t)t * T_ + s] - mx);
  red[tid] = sum; __syncthreads();
  for (int off = 128; off > 0; off >>= 1) {
    if (tid < off) red[tid] += red[tid + off];
    __syncthreads();
  }
  const float inv = 1.f / red[0];

  for (int t = tid; t < T_; t += 256)
    P[(size_t)t * T_ + s] = (__bf16)(__expf(S[(size_t)t * T_ + s] - mx) * inv);
}

// O = P[T,T] x V ; V supplied transposed Vt[E,T] -> B^T row-major directly.
// Store into Hcat[(t*B+b)][hh*E+e] bf16.
__global__ __launch_bounds__(128) void k_ogemm(const __bf16* __restrict__ P,
                                               const __bf16* __restrict__ Vt,
                                               __bf16* __restrict__ Hcat,
                                               int hh, int b) {
  int M0, N0; wave_origin(M0, N0);   // M over t, N over e
  Acc2x4 acc = {};
  gemm_core_2x4(P + (size_t)M0 * T_, T_, Vt + (size_t)N0 * T_, T_, T_, acc);

  const int lane = threadIdx.x & 31;
  const int n = lane & 15, h = lane >> 4;
#pragma unroll
  for (int ti = 0; ti < 2; ++ti)
#pragma unroll
    for (int tj = 0; tj < 4; ++tj)
#pragma unroll
      for (int r = 0; r < 8; ++r) {
        const int t = M0 + ti * 16 + h * 8 + r;
        const int e = N0 + tj * 16 + n;
        Hcat[(size_t)(t * B_ + b) * HE + hh * E_ + e] = (__bf16)acc.m[ti][tj][r];
      }
}

// out = Hcat[NROW,HE] x Wo[D,HE]^T + bo -> fp32 [NROW,D] (== [T,B,D] flat)
__global__ __launch_bounds__(128) void k_outproj(const __bf16* __restrict__ Hcat,
                                                 const __bf16* __restrict__ Wo,
                                                 const float* __restrict__ bo,
                                                 float* __restrict__ out) {
  int M0, N0; wave_origin(M0, N0);
  Acc2x4 acc = {};
  gemm_core_2x4(Hcat + (size_t)M0 * HE, HE, Wo + (size_t)N0 * HE, HE, HE, acc);

  const int lane = threadIdx.x & 31;
  const int n = lane & 15, h = lane >> 4;
#pragma unroll
  for (int ti = 0; ti < 2; ++ti)
#pragma unroll
    for (int tj = 0; tj < 4; ++tj)
#pragma unroll
      for (int r = 0; r < 8; ++r) {
        const int row = M0 + ti * 16 + h * 8 + r;
        const int col = N0 + tj * 16 + n;
        out[(size_t)row * D_ + col] = acc.m[ti][tj][r] + bo[col];
      }
}

// ---------------- launcher -------------------------------------------------

extern "C" void kernel_launch(void* const* d_in, const int* in_sizes, int n_in,
                              void* d_out, int out_size, void* d_ws, size_t ws_size,
                              hipStream_t stream) {
  const float* Q  = (const float*)d_in[0];
  const float* Wq = (const float*)d_in[1];
  const float* bq = (const float*)d_in[2];
  const float* Wk = (const float*)d_in[3];
  const float* bk = (const float*)d_in[4];
  const float* Wv = (const float*)d_in[5];
  const float* bv = (const float*)d_in[6];
  const float* Wo = (const float*)d_in[7];
  const float* bo = (const float*)d_in[8];
  float* out = (float*)d_out;

  char* ws = (char*)d_ws;
  size_t off = 0;
  auto carve = [&](size_t bytes) {
    void* p = ws + off;
    off += (bytes + 255) & ~(size_t)255;
    return p;
  };
  __bf16* Xb   = (__bf16*)carve((size_t)NROW * D_ * 2);
  __bf16* Wqb  = (__bf16*)carve((size_t)HE * D_ * 2);
  __bf16* Wkb  = (__bf16*)carve((size_t)HE * D_ * 2);
  __bf16* Wvb  = (__bf16*)carve((size_t)HE * D_ * 2);
  __bf16* Wob  = (__bf16*)carve((size_t)D_ * HE * 2);
  __bf16* Qp   = (__bf16*)carve((size_t)H_ * B_ * T_ * E_ * 2);  // [h][b][t][e]
  __bf16* Kp   = (__bf16*)carve((size_t)H_ * B_ * T_ * E_ * 2);  // [h][b][t][e]
  __bf16* Vt   = (__bf16*)carve((size_t)H_ * B_ * E_ * T_ * 2);  // [h][b][e][t]
  __bf16* Hcat = (__bf16*)carve((size_t)NROW * HE * 2);          // [t*B+b][h*E+e]
  float*  S    = (float*) carve((size_t)T_ * T_ * 4);            // per-slice, reused
  __bf16* P    = (__bf16*)carve((size_t)T_ * T_ * 2);            // per-slice, reused

  // 1) casts fp32 -> bf16
  k_cast<<<512, 256, 0, stream>>>(Q,  Xb,  NROW * D_);
  k_cast<<<512, 256, 0, stream>>>(Wq, Wqb, HE * D_);
  k_cast<<<512, 256, 0, stream>>>(Wk, Wkb, HE * D_);
  k_cast<<<512, 256, 0, stream>>>(Wv, Wvb, HE * D_);
  k_cast<<<512, 256, 0, stream>>>(Wo, Wob, D_ * HE);

  // 2) QKV projections (1/sqrt(E) folded into Q, V stored transposed)
  const float scale = 0.044194173824159216f;  // 1/sqrt(512)
  dim3 gq(NROW / 64, HE / 128);
  k_qkv<<<gq, 128, 0, stream>>>(Xb, Wqb, bq, Qp, scale, 0);
  k_qkv<<<gq, 128, 0, stream>>>(Xb, Wkb, bk, Kp, 1.0f, 0);
  k_qkv<<<gq, 128, 0, stream>>>(Xb, Wvb, bv, Vt, 1.0f, 1);

  // 3) attention, one (h,b) slice at a time (stream-serialized, S/P reused)
  for (int h = 0; h < H_; ++h)
    for (int b = 0; b < B_; ++b) {
      const size_t slice = (size_t)(h * B_ + b);
      const __bf16* Qhb = Qp + slice * T_ * E_;
      const __bf16* Khb = Kp + slice * T_ * E_;
      const __bf16* Vhb = Vt + slice * E_ * T_;
      k_scores<<<dim3(T_ / 64, T_ / 128), 128, 0, stream>>>(Qhb, Khb, S);
      k_softmax_col<<<T_, 256, 0, stream>>>(S, P);
      k_ogemm<<<dim3(T_ / 64, E_ / 128), 128, 0, stream>>>(P, Vhb, Hcat, h, b);
    }

  // 4) output projection
  k_outproj<<<dim3(NROW / 64, D_ / 128), 128, 0, stream>>>(Hcat, Wob, bo, out);
}
